// TransformerBlock_55748675502131
// MI455X (gfx1250) — matrix operand order repaired and checked by reading, not compiled, not run on hardware
//
#include <hip/hip_runtime.h>
#include <hip/hip_bf16.h>
#include <math.h>

// MI455X / gfx1250 implementation of the Restormer-style TransformerBlock.
// Big 1x1-conv GEMMs + q.k^T run on v_wmma_f32_16x16x32_f16 (wave32 WMMA),
// intermediates kept in f16. GEMMs are register-blocked 4x over M so each
// streaming B-fragment feeds 4 WMMAs. Requires ws_size >= ~560 MB.

typedef __attribute__((ext_vector_type(16))) _Float16 v16h;
typedef __attribute__((ext_vector_type(8)))  _Float16 v8h;
typedef __attribute__((ext_vector_type(8)))  float    v8f;

#define DIMC 192
#define C3   576
#define HIDC 576
#define BATCH 8
#define WDIM 128
#define HWSZ 16384                 // 128*128
#define NPIX (BATCH * HWSZ)        // 131072

// ---------------------------------------------------------------- utilities
__global__ void cvt_f16_kernel(const float* __restrict__ src,
                               _Float16* __restrict__ dst, int n) {
  int i = blockIdx.x * 256 + threadIdx.x;
  if (i < n) dst[i] = (_Float16)src[i];
}

// ------------------------------------------------------------- LayerNorm
// Block = 256 threads = 64 pixels x 4 channel-slices (48 ch each).
// NCHW_OUT=0: write pixel-major f16 [n,192] via LDS tile (coalesced uint4).
// NCHW_OUT=1: write f16 NCHW directly.
template <int NCHW_OUT>
__global__ void ln_kernel(const float* __restrict__ x, const float* __restrict__ g,
                          const float* __restrict__ beta, _Float16* __restrict__ out) {
  __shared__ float s_sum[4][64];
  __shared__ float s_sq[4][64];
  __shared__ float s_mu[64];
  __shared__ float s_inv[64];
  __shared__ _Float16 tile[64 * DIMC];

  int tid = threadIdx.x;
  int p = tid & 63, cs = tid >> 6;
  long n = (long)blockIdx.x * 64 + p;
  int b = (int)(n >> 14), hw = (int)(n & 16383);
  const float* xb = x + ((long)b * DIMC) * HWSZ + hw;

  float sum = 0.f, sq = 0.f;
  for (int j = 0; j < 48; ++j) {
    float v = xb[(long)(cs * 48 + j) * HWSZ];
    sum += v; sq += v * v;
  }
  s_sum[cs][p] = sum; s_sq[cs][p] = sq;
  __syncthreads();
  if (cs == 0) {
    float ts = s_sum[0][p] + s_sum[1][p] + s_sum[2][p] + s_sum[3][p];
    float tq = s_sq[0][p] + s_sq[1][p] + s_sq[2][p] + s_sq[3][p];
    float mu = ts * (1.f / DIMC);
    float var = tq * (1.f / DIMC) - mu * mu;
    s_mu[p] = mu; s_inv[p] = rsqrtf(var + 1e-5f);
  }
  __syncthreads();
  float mu = s_mu[p], inv = s_inv[p];

  if (NCHW_OUT) {
    _Float16* ob = out + ((long)b * DIMC) * HWSZ + hw;
    for (int j = 0; j < 48; ++j) {
      int c = cs * 48 + j;
      float v = (xb[(long)c * HWSZ] - mu) * inv * g[c] + beta[c];
      ob[(long)c * HWSZ] = (_Float16)v;
    }
  } else {
    for (int j = 0; j < 48; ++j) {
      int c = cs * 48 + j;
      float v = (xb[(long)c * HWSZ] - mu) * inv * g[c] + beta[c];
      tile[p * DIMC + c] = (_Float16)v;
    }
    __syncthreads();
    // 64*192 f16 = 24 KB contiguous region per block; flush as uint4.
    const uint4* ts4 = (const uint4*)tile;
    uint4* od4 = (uint4*)(out + (long)blockIdx.x * 64 * DIMC);
    for (int i = tid; i < 64 * DIMC / 8; i += 256) od4[i] = ts4[i];
  }
}

// ------------------------------------------------------------- WMMA GEMM
// C[M, NPIX] = W[M,K](f16 row-major) x Y[NPIX,K]^T (f16 pixel-major).
// Block = 8 waves along N; each wave owns a 64(M) x 16(N) strip as 4
// accumulator fragments, so each B fragment load feeds 4 WMMAs.
// A-frag (16x32 f16): lane row = lane%16, 16 contiguous K at 16h;
// B-frag: lane col = lane%16, 16 contiguous K at 16h.
// MODE 0: +bias -> f16 NCHW.  MODE 1: +bias +extra(f32 NCHW) -> f32 NCHW.
// MODE 2: +bias, exact GELU -> f16 NCHW.
template <int MODE>
__global__ void gemm_wmma(const _Float16* __restrict__ W, const _Float16* __restrict__ Y,
                          const float* __restrict__ bias, const float* __restrict__ extra,
                          void* __restrict__ outv, int M, int K) {
  int tid = threadIdx.x;
  int lane = tid & 31, wave = tid >> 5;
  int l16 = lane & 15, h = lane >> 4;
  int m0 = blockIdx.y * 64;                     // 4 M-tiles per wave
  long n0 = ((long)blockIdx.x * 8 + wave) * 16;

  const _Float16* wrow0 = W + (long)(m0 + l16) * K;
  const _Float16* yrow  = Y + (n0 + l16) * K;
  v8f acc0 = {}, acc1 = {}, acc2 = {}, acc3 = {};

  for (int k0 = 0; k0 < K; k0 += 32) {
    __builtin_prefetch(yrow + k0 + 128, 0, 0);  // global_prefetch_b8
    v16h bm = *(const v16h*)(yrow + k0 + 16 * h);
#pragma unroll
    for (int t = 0; t < 4; ++t) {
      const _Float16* wr = wrow0 + (long)t * 16 * K;
      v16h a = *(const v16h*)(wr + k0 + 16 * h);
      if (t == 0)
        acc0 = __builtin_amdgcn_wmma_f32_16x16x32_f16(false, a, false, bm,
                                                      (short)0, acc0, false, false);
      else if (t == 1)
        acc1 = __builtin_amdgcn_wmma_f32_16x16x32_f16(false, a, false, bm,
                                                      (short)0, acc1, false, false);
      else if (t == 2)
        acc2 = __builtin_amdgcn_wmma_f32_16x16x32_f16(false, a, false, bm,
                                                      (short)0, acc2, false, false);
      else
        acc3 = __builtin_amdgcn_wmma_f32_16x16x32_f16(false, a, false, bm,
                                                      (short)0, acc3, false, false);
    }
  }

  long n = n0 + l16;
  int b = (int)(n >> 14);
  int hw = (int)(n & 16383);
#pragma unroll
  for (int t = 0; t < 4; ++t) {
    v8f acc = (t == 0) ? acc0 : (t == 1) ? acc1 : (t == 2) ? acc2 : acc3;
#pragma unroll
    for (int v = 0; v < 8; ++v) {
      int m = m0 + t * 16 + v + 8 * h;
      float val = acc[v] + bias[m];
      long off = ((long)b * M + m) * HWSZ + hw;
      if (MODE == 0) {
        ((_Float16*)outv)[off] = (_Float16)val;
      } else if (MODE == 1) {
        ((float*)outv)[off] = val + extra[off];
      } else {
        float ge = 0.5f * val * (1.f + erff(val * 0.70710678f));
        ((_Float16*)outv)[off] = (_Float16)ge;
      }
    }
  }
}

// --------------------------------------------------- 3x3 depthwise/grouped
// in: f16 NCHW [B,Cin,HW]. out channel o uses in channel o*Cin/Cout.
// prompt (192 floats) optionally added (qkv path). pixMajor selects output
// layout: [n,Cout] f16 (feeds next GEMM) or NCHW f16 (feeds next conv).
__global__ void dwconv3_kernel(const _Float16* __restrict__ in, const float* __restrict__ wt,
                               const float* __restrict__ bias, const float* __restrict__ prompt,
                               _Float16* __restrict__ out, int Cin, int Cout, int pixMajor) {
  long idx = (long)blockIdx.x * 256 + threadIdx.x;   // B*Cout*HW threads
  int hw = (int)(idx & 16383);
  long t = idx >> 14;
  int o = (int)(t % Cout);
  int b = (int)(t / Cout);
  int ic = (int)(((long)o * Cin) / Cout);
  int hh = hw >> 7, ww = hw & 127;
  const _Float16* ip = in + ((long)b * Cin + ic) * HWSZ;
  const float* w9 = wt + o * 9;
  float acc = bias[o];
#pragma unroll
  for (int dh = -1; dh <= 1; ++dh) {
    int h2 = hh + dh;
    if (h2 < 0 || h2 >= WDIM) continue;
#pragma unroll
    for (int dw = -1; dw <= 1; ++dw) {
      int w2 = ww + dw;
      if (w2 < 0 || w2 >= WDIM) continue;
      acc += w9[(dh + 1) * 3 + (dw + 1)] * (float)ip[h2 * WDIM + w2];
    }
  }
  if (prompt) acc += prompt[o % DIMC];
  if (pixMajor) out[((long)b * HWSZ + hw) * Cout + o] = (_Float16)acc;
  else          out[((long)b * Cout + o) * HWSZ + hw] = (_Float16)acc;
}

// ------------------------------------------- L2 row norms for q,k (ch 0..383)
__global__ void rownorm_kernel(const _Float16* __restrict__ qkv, float* __restrict__ invqk) {
  __shared__ float red[256];
  int b = blockIdx.x / 384, cc = blockIdx.x % 384;
  const _Float16* row = qkv + ((long)b * C3 + cc) * HWSZ;
  float s = 0.f;
  for (int i = threadIdx.x; i < HWSZ; i += 256) {
    float v = (float)row[i];
    s += v * v;
  }
  red[threadIdx.x] = s;
  __syncthreads();
  for (int st = 128; st > 0; st >>= 1) {
    if (threadIdx.x < st) red[threadIdx.x] += red[threadIdx.x + st];
    __syncthreads();
  }
  if (threadIdx.x == 0) invqk[b * 384 + cc] = 1.f / fmaxf(sqrtf(red[0]), 1e-12f);
}

// ----------------------------------------- attn = (q/|q|)(k/|k|)^T * temp
// grid (9 tiles, 32 b*head). 8 waves split K=16384 (2048 each), WMMA
// accumulate, LDS-reduce the 8 partial 16x16 tiles, scale in epilogue.
__global__ void attn_qk_kernel(const _Float16* __restrict__ qkv, const float* __restrict__ invqk,
                               const float* __restrict__ temp, float* __restrict__ attn) {
  __shared__ float red[8 * 256];
  int tid = threadIdx.x, lane = tid & 31, wave = tid >> 5;
  int l16 = lane & 15, h = lane >> 4;
  int tl = blockIdx.x;           // tile 0..8  (3x3 of 16 over 48x48)
  int bh = blockIdx.y;           // 0..31
  int b = bh >> 2, head = bh & 3;
  int tm = tl / 3, tn = tl % 3;
  const _Float16* qrow = qkv + ((long)b * C3 + head * 48 + tm * 16 + l16) * HWSZ;
  const _Float16* krow = qkv + ((long)b * C3 + 192 + head * 48 + tn * 16 + l16) * HWSZ;

  v8f acc = {};
  int kbase = wave * 2048;
  for (int k0 = kbase; k0 < kbase + 2048; k0 += 32) {
    v16h a = *(const v16h*)(qrow + k0 + 16 * h);
    v16h bm = *(const v16h*)(krow + k0 + 16 * h);
    acc = __builtin_amdgcn_wmma_f32_16x16x32_f16(false, a, false, bm,
                                                 (short)0, acc, false, false);
  }
#pragma unroll
  for (int v = 0; v < 8; ++v) red[wave * 256 + 32 * v + lane] = acc[v];
  __syncthreads();

  int lp = tid & 31, vv = tid >> 5;      // each thread owns one (v, lane) slot
  float s = 0.f;
#pragma unroll
  for (int w = 0; w < 8; ++w) s += red[w * 256 + 32 * vv + lp];
  int mrow = tm * 16 + vv + 8 * (lp >> 4);
  int ncol = tn * 16 + (lp & 15);
  float val = s * invqk[b * 384 + head * 48 + mrow]
                * invqk[b * 384 + 192 + head * 48 + ncol] * temp[head];
  attn[((long)bh * 48 + mrow) * 48 + ncol] = val;
}

// --------------------------------------------------------- softmax over 48
__global__ void softmax48_kernel(float* __restrict__ attn) {
  int bh = blockIdx.x, r = threadIdx.x;
  if (r >= 48) return;
  float* row = attn + ((long)bh * 48 + r) * 48;
  float mx = row[0];
  for (int i = 1; i < 48; ++i) mx = fmaxf(mx, row[i]);
  float e[48], s = 0.f;
  for (int i = 0; i < 48; ++i) { e[i] = expf(row[i] - mx); s += e[i]; }
  float inv = 1.f / s;
  for (int i = 0; i < 48; ++i) row[i] = e[i] * inv;
}

// ------------------------------------ out = attn @ v  -> pixel-major f16
// K=48 is too small/strided for WMMA B-frags; VALU with attn in LDS.
__global__ void attn_v_kernel(const float* __restrict__ attn, const _Float16* __restrict__ qkv,
                              _Float16* __restrict__ outPM) {
  __shared__ float A[48 * 48];
  int bh = blockIdx.y, b = bh >> 2, head = bh & 3;
  const float* ap = attn + (long)bh * 48 * 48;
  for (int i = threadIdx.x; i < 2304; i += 256) A[i] = ap[i];
  __syncthreads();
  int hw = blockIdx.x * 256 + threadIdx.x;
  const _Float16* vbase = qkv + ((long)b * C3 + 384 + head * 48) * HWSZ + hw;
  float vv[48];
#pragma unroll
  for (int d = 0; d < 48; ++d) vv[d] = (float)vbase[(long)d * HWSZ];
  _Float16* ob = outPM + ((long)b * HWSZ + hw) * DIMC + head * 48;
  for (int c = 0; c < 48; ++c) {
    float s = 0.f;
#pragma unroll
    for (int d = 0; d < 48; ++d) s += A[c * 48 + d] * vv[d];
    ob[c] = (_Float16)s;
  }
}

// ================================================================ launcher
extern "C" void kernel_launch(void* const* d_in, const int* in_sizes, int n_in,
                              void* d_out, int out_size, void* d_ws, size_t ws_size,
                              hipStream_t stream) {
  (void)in_sizes; (void)n_in; (void)out_size; (void)ws_size;
  const float* x         = (const float*)d_in[0];
  const float* ln1_w     = (const float*)d_in[1];
  const float* ln1_b     = (const float*)d_in[2];
  const float* qkv_w     = (const float*)d_in[3];
  const float* qkv_b     = (const float*)d_in[4];
  const float* qkv_dw_w  = (const float*)d_in[5];
  const float* qkv_dw_b  = (const float*)d_in[6];
  const float* temper    = (const float*)d_in[7];
  const float* prompt    = (const float*)d_in[8];
  const float* proj_w    = (const float*)d_in[9];
  const float* proj_b    = (const float*)d_in[10];
  const float* ln2_w     = (const float*)d_in[11];
  const float* ln2_b     = (const float*)d_in[12];
  const float* ffn_dw1_w = (const float*)d_in[13];
  const float* ffn_dw1_b = (const float*)d_in[14];
  const float* ffn_pm_w  = (const float*)d_in[15];
  const float* ffn_pm_b  = (const float*)d_in[16];
  const float* ffn_dw2_w = (const float*)d_in[17];
  const float* ffn_dw2_b = (const float*)d_in[18];
  const float* ffn_po_w  = (const float*)d_in[19];
  const float* ffn_po_b  = (const float*)d_in[20];

  char* ws = (char*)d_ws;
  _Float16* bufP  = (_Float16*)(ws);                        // [n,K] pixel-major, 151MB
  _Float16* bufN1 = (_Float16*)(ws + 0x09000000);           // f16 NCHW 576ch, 151MB
  _Float16* bufN2 = (_Float16*)(ws + 0x12000000);           // f16 NCHW 576ch, 151MB
  float*    xres  = (float*)   (ws + 0x1B000000);           // f32 NCHW 192ch, 101MB
  _Float16* wbuf  = (_Float16*)(ws + 0x21000000);           // f16 weights, 1.2MB
  float*    attnb = (float*)   (ws + 0x21200000);           // [32][48][48] f32
  float*    invqk = (float*)   (ws + 0x21300000);           // [8][384] f32

  _Float16* wqkv = wbuf;
  _Float16* wprj = wbuf + 110592;
  _Float16* wpm  = wbuf + 147456;
  _Float16* wpo  = wbuf + 479232;

  // weight conversion f32 -> f16
  cvt_f16_kernel<<<(110592 + 255) / 256, 256, 0, stream>>>(qkv_w,    wqkv, 110592);
  cvt_f16_kernel<<<(36864  + 255) / 256, 256, 0, stream>>>(proj_w,   wprj, 36864);
  cvt_f16_kernel<<<(331776 + 255) / 256, 256, 0, stream>>>(ffn_pm_w, wpm,  331776);
  cvt_f16_kernel<<<(110592 + 255) / 256, 256, 0, stream>>>(ffn_po_w, wpo,  110592);

  // ---- attention branch ----
  ln_kernel<0><<<NPIX / 64, 256, 0, stream>>>(x, ln1_w, ln1_b, bufP);
  gemm_wmma<0><<<dim3(NPIX / 128, C3 / 64), 256, 0, stream>>>(
      wqkv, bufP, qkv_b, nullptr, bufN1, C3, DIMC);
  dwconv3_kernel<<<BATCH * C3 * 64, 256, 0, stream>>>(
      bufN1, qkv_dw_w, qkv_dw_b, prompt, bufN2, C3, C3, 0);
  rownorm_kernel<<<BATCH * 384, 256, 0, stream>>>(bufN2, invqk);
  attn_qk_kernel<<<dim3(9, 32), 256, 0, stream>>>(bufN2, invqk, temper, attnb);
  softmax48_kernel<<<32, 64, 0, stream>>>(attnb);
  attn_v_kernel<<<dim3(HWSZ / 256, 32), 256, 0, stream>>>(attnb, bufN2, bufP);
  gemm_wmma<1><<<dim3(NPIX / 128, DIMC / 64), 256, 0, stream>>>(
      wprj, bufP, proj_b, x, xres, DIMC, DIMC);

  // ---- FFN branch ----
  ln_kernel<1><<<NPIX / 64, 256, 0, stream>>>(xres, ln2_w, ln2_b, bufN1);
  dwconv3_kernel<<<BATCH * HIDC * 64, 256, 0, stream>>>(
      bufN1, ffn_dw1_w, ffn_dw1_b, nullptr, bufP, DIMC, HIDC, 1);
  gemm_wmma<2><<<dim3(NPIX / 128, HIDC / 64), 256, 0, stream>>>(
      wpm, bufP, ffn_pm_b, nullptr, bufN2, HIDC, HIDC);
  dwconv3_kernel<<<BATCH * HIDC * 64, 256, 0, stream>>>(
      bufN2, ffn_dw2_w, ffn_dw2_b, nullptr, bufP, HIDC, HIDC, 1);
  gemm_wmma<1><<<dim3(NPIX / 128, DIMC / 64), 256, 0, stream>>>(
      wpo, bufP, ffn_po_b, xres, (float*)d_out, DIMC, HIDC);
}
